// LemLightV2_30477087933090
// MI455X (gfx1250) — compile-verified, matrix operand order checked by reading, not run on hardware
//
#include <hip/hip_runtime.h>

typedef _Float16 half8  __attribute__((ext_vector_type(8)));
typedef _Float16 half16 __attribute__((ext_vector_type(16)));
typedef float    f32x8  __attribute__((ext_vector_type(8)));

// f16 weight scratch layout in d_ws (all transposed to [N][K] so a B-fragment
// for v_wmma_f32_16x16x32_f16 is two contiguous 16-byte reads per lane):
//   W1T : [128][160]  (K padded 136 -> 160)   offset 0      (20480 halves)
//   W2T : [128][128]                          offset 20480
//   W3T : [128][128]                          offset 36864
//   WEnvT: [96][128]                          offset 53248
#define W1T_OFF   0
#define W2T_OFF   20480
#define W3T_OFF   36864
#define WENV_OFF  53248
#define WTS_TOTAL 65536

__device__ __forceinline__ half16 cat88(half8 lo, half8 hi) {
  half16 r;
#pragma unroll
  for (int i = 0; i < 8; ++i) { r[i] = lo[i]; r[i + 8] = hi[i]; }
  return r;
}
// Fragment load: halves {K..K+7} and {K+16..K+23} relative to a 32-wide K chunk.
// Low half-wave (lanes 0-15) takes K base 0, high half-wave takes K base 8,
// matching the 16-bit A/B operand layout in the CDNA5 WMMA spec.
__device__ __forceinline__ half16 ldfrag(const _Float16* p) {
  half8 lo = *(const half8*)(p);
  half8 hi = *(const half8*)(p + 16);
  return cat88(lo, hi);
}

// silu with hardware reciprocal (v_rcp_f32) instead of IEEE divide
__device__ __forceinline__ float silu(float x) {
  return x * __builtin_amdgcn_rcpf(1.0f + __expf(-x));
}

// ---------------------------------------------------------------- weights f32 -> f16 (transposed, padded)
__global__ void convert_weights(const float* __restrict__ W1, const float* __restrict__ W2,
                                const float* __restrict__ W3, const float* __restrict__ We,
                                _Float16* __restrict__ wts) {
  int i = blockIdx.x * blockDim.x + threadIdx.x;
  if (i >= WTS_TOTAL) return;
  if (i < W2T_OFF) {                       // W1T [128][160], W1 is [136][128]
    int n = i / 160, k = i % 160;
    wts[i] = (k < 136) ? (_Float16)W1[k * 128 + n] : (_Float16)0.0f;
  } else if (i < W3T_OFF) {                // W2T [128][128]
    int j = i - W2T_OFF; int n = j / 128, k = j % 128;
    wts[i] = (_Float16)W2[k * 128 + n];
  } else if (i < WENV_OFF) {               // W3T [128][128]
    int j = i - W3T_OFF; int n = j / 128, k = j % 128;
    wts[i] = (_Float16)W3[k * 128 + n];
  } else {                                 // WEnvT [96][128], W_env is [128][96]
    int j = i - WENV_OFF; int n = j / 128, k = j % 128;
    wts[i] = (_Float16)We[k * 96 + n];
  }
}

// ---------------------------------------------------------------- main edge kernel (WMMA MLP)
__global__ void __launch_bounds__(256)
edge_kernel(const int* __restrict__ edge_index, const float* __restrict__ edge_sh,
            const float* __restrict__ edge_length, const float* __restrict__ edge_one_hot,
            const float* __restrict__ bessel_w, const _Float16* __restrict__ wts,
            float* __restrict__ out_lat, float* __restrict__ node_accum,
            float* __restrict__ out_ef, float* __restrict__ out_cut,
            float* __restrict__ out_mask, int E, int numTiles) {
  __shared__ __align__(16) _Float16 smem[8][16][160]; // per-wave 16-edge activation tile
  __shared__ float cutb[8][16];
  __shared__ float shb[8][16][9];
  __shared__ int   srcb[8][16];

  const int wave = threadIdx.x >> 5;
  const int lane = threadIdx.x & 31;
  const int ln15 = lane & 15;   // A-row / B-col / D-col owned by this lane
  const int half = lane >> 4;   // selects K sub-range of the 32-wide chunk

  for (int tg = blockIdx.x; tg * 8 < numTiles; tg += gridDim.x) {
    const int t = tg * 8 + wave;
    const bool act = (t < numTiles);
    const long long ebase = (long long)t * 16;

    // ---- stage 1: stage one_hot (f16) + bessel + cutoff + sh + src into LDS
    if (act) {
      long long e = ebase + ln15;
      long long eg = (e < E) ? e : (long long)E - 1;
      const float4* src4 = (const float4*)(edge_one_hot + eg * 128 + half * 64);
#pragma unroll
      for (int j = 0; j < 16; ++j) {
        float4 v = src4[j];
        int c = half * 64 + j * 4;
        smem[wave][ln15][c + 0] = (_Float16)v.x;
        smem[wave][ln15][c + 1] = (_Float16)v.y;
        smem[wave][ln15][c + 2] = (_Float16)v.z;
        smem[wave][ln15][c + 3] = (_Float16)v.w;
      }
      if (half == 0) {
        float r  = edge_length[eg];
        float x  = r * (1.0f / 5.0f);
        float x2 = x * x, x3 = x2 * x, x6 = x3 * x3;
        float cut = (x < 1.0f) ? (1.0f - 28.0f * x6 + 48.0f * x6 * x - 21.0f * x6 * x2) : 0.0f;
        if (e >= E) cut = 0.0f;
        cutb[wave][ln15] = cut;
        float inv_r = __builtin_amdgcn_rcpf(r);
        float amp = sqrtf(2.0f / 5.0f);
#pragma unroll
        for (int j = 0; j < 8; ++j)
          smem[wave][ln15][128 + j] = (_Float16)(amp * __sinf(bessel_w[j] * x) * inv_r);
#pragma unroll
        for (int j = 0; j < 9; ++j) shb[wave][ln15][j] = edge_sh[eg * 9 + j];
        srcb[wave][ln15] = edge_index[eg];
        if (e < E) {
          out_cut[e]  = cut;
          out_mask[e] = (cut > 0.0f) ? 1.0f : 0.0f;
        }
      } else {
#pragma unroll
        for (int j = 0; j < 24; ++j) smem[wave][ln15][136 + j] = (_Float16)0.0f; // K pad
      }
    }
    __syncthreads();

    // ---- layer 1: [16x160] @ W1T -> silu -> LDS (K chunks = 5)
    half16 a1[5];
    if (act) {
#pragma unroll
      for (int c = 0; c < 5; ++c)
        a1[c] = ldfrag(&smem[wave][ln15][c * 32 + half * 8]);
    }
    __syncthreads();
    if (act) {
#pragma unroll
      for (int n = 0; n < 8; ++n) {
        half16 b[5];
#pragma unroll
        for (int c = 0; c < 5; ++c)  // batched: one s_clause, one wait
          b[c] = ldfrag(wts + W1T_OFF + (16 * n + ln15) * 160 + c * 32 + half * 8);
        f32x8 acc = {};
#pragma unroll
        for (int c = 0; c < 5; ++c)  // back-to-back WMMA chain
          acc = __builtin_amdgcn_wmma_f32_16x16x32_f16(false, a1[c], false, b[c],
                                                       (short)0, acc, false, false);
#pragma unroll
        for (int r = 0; r < 8; ++r)
          smem[wave][r + half * 8][16 * n + ln15] = (_Float16)silu(acc[r]);
      }
    }
    __syncthreads();

    // ---- layer 2: [16x128] @ W2T -> silu -> LDS (K chunks = 4)
    half16 a2[4];
    if (act) {
#pragma unroll
      for (int c = 0; c < 4; ++c)
        a2[c] = ldfrag(&smem[wave][ln15][c * 32 + half * 8]);
    }
    __syncthreads();
    if (act) {
#pragma unroll
      for (int n = 0; n < 8; ++n) {
        half16 b[4];
#pragma unroll
        for (int c = 0; c < 4; ++c)
          b[c] = ldfrag(wts + W2T_OFF + (16 * n + ln15) * 128 + c * 32 + half * 8);
        f32x8 acc = {};
#pragma unroll
        for (int c = 0; c < 4; ++c)
          acc = __builtin_amdgcn_wmma_f32_16x16x32_f16(false, a2[c], false, b[c],
                                                       (short)0, acc, false, false);
#pragma unroll
        for (int r = 0; r < 8; ++r)
          smem[wave][r + half * 8][16 * n + ln15] = (_Float16)silu(acc[r]);
      }
    }
    __syncthreads();

    // ---- layer 3: latents = cut * (h @ W3T); write latents, keep f16 copy in LDS
    half16 a3[4];
    if (act) {
#pragma unroll
      for (int c = 0; c < 4; ++c)
        a3[c] = ldfrag(&smem[wave][ln15][c * 32 + half * 8]);
    }
    __syncthreads();
    if (act) {
#pragma unroll
      for (int n = 0; n < 8; ++n) {
        half16 b[4];
#pragma unroll
        for (int c = 0; c < 4; ++c)
          b[c] = ldfrag(wts + W3T_OFF + (16 * n + ln15) * 128 + c * 32 + half * 8);
        f32x8 acc = {};
#pragma unroll
        for (int c = 0; c < 4; ++c)
          acc = __builtin_amdgcn_wmma_f32_16x16x32_f16(false, a3[c], false, b[c],
                                                       (short)0, acc, false, false);
#pragma unroll
        for (int r = 0; r < 8; ++r) {
          int Mrow = r + half * 8;
          float lat = acc[r] * cutb[wave][Mrow];
          long long e = ebase + Mrow;
          if (e < E) out_lat[e * 128 + 16 * n + ln15] = lat;
          smem[wave][Mrow][16 * n + ln15] = (_Float16)lat;
        }
      }
    }
    __syncthreads();

    // ---- layer 4: w = latents @ WEnvT (96 cols), expand by spherical harmonics,
    //              write edge_features + atomic segment-sum into node accumulator
    half16 a4[4];
    if (act) {
#pragma unroll
      for (int c = 0; c < 4; ++c)
        a4[c] = ldfrag(&smem[wave][ln15][c * 32 + half * 8]);
    }
    if (act) {
#pragma unroll
      for (int n = 0; n < 6; ++n) {
        half16 b[4];
#pragma unroll
        for (int c = 0; c < 4; ++c)
          b[c] = ldfrag(wts + WENV_OFF + (16 * n + ln15) * 128 + c * 32 + half * 8);
        f32x8 acc = {};
#pragma unroll
        for (int c = 0; c < 4; ++c)
          acc = __builtin_amdgcn_wmma_f32_16x16x32_f16(false, a4[c], false, b[c],
                                                       (short)0, acc, false, false);
        const int col  = 16 * n + ln15;   // 0..95
        const int lidx = col >> 5;        // which angular block (d = 1,3,5)
        const int m    = col & 31;
        int ofc, sho, d;
        if (lidx == 0)      { ofc = m;            sho = 0; d = 1; }
        else if (lidx == 1) { ofc = 32 + m * 3;   sho = 1; d = 3; }
        else                { ofc = 128 + m * 5;  sho = 4; d = 5; }
#pragma unroll
        for (int r = 0; r < 8; ++r) {
          int Mrow = r + half * 8;
          long long e = ebase + Mrow;
          if (e >= E) continue;
          float mk = (cutb[wave][Mrow] > 0.0f) ? 1.0f : 0.0f;
          float wv = acc[r] * mk;
          int srcn = srcb[wave][Mrow];
          for (int j = 0; j < d; ++j) {
            float v = wv * shb[wave][Mrow][sho + j];
            out_ef[e * 288 + ofc + j] = v;
            atomicAdd(&node_accum[(long long)srcn * 288 + ofc + j], v);
          }
        }
      }
    }
    __syncthreads(); // protect LDS before next iteration's stage 1
  }
}

// ---------------------------------------------------------------- node layer-norm (in place)
__global__ void node_kernel(float* __restrict__ node, const float* __restrict__ gamma,
                            const float* __restrict__ beta, int NN) {
  int i = blockIdx.x * blockDim.x + threadIdx.x;
  if (i >= NN) return;
  float* p = node + (long long)i * 288;
  const float sc = 0.22360679774997896f; // 20^-0.5
  float s[32];
  float mean = 0.0f;
#pragma unroll
  for (int k = 0; k < 32; ++k) { s[k] = p[k] * sc; mean += s[k]; }
  mean *= (1.0f / 32.0f);
  float ss = 0.0f;
#pragma unroll
  for (int k = 0; k < 32; ++k) { s[k] -= mean; ss += s[k] * s[k]; }
  float sv1 = 0.0f;
  for (int k = 32; k < 128; ++k) { float v = p[k] * sc; sv1 += v * v; }
  float sv2 = 0.0f;
  for (int k = 128; k < 288; ++k) { float v = p[k] * sc; sv2 += v * v; }
  float var = (ss * (1.0f / 3.0f) + sv1 * (1.0f / 9.0f) + sv2 * (1.0f / 15.0f)) * (1.0f / 32.0f);
  float inv = rsqrtf(var + 1e-8f);
#pragma unroll
  for (int k = 0; k < 32; ++k) p[k] = s[k] * inv * gamma[k] + beta[k];
  for (int m = 0; m < 32; ++m) {
    float g = inv * gamma[32 + m] * sc;
    for (int j = 0; j < 3; ++j) { int idx = 32 + m * 3 + j; p[idx] = p[idx] * g; }
  }
  for (int m = 0; m < 32; ++m) {
    float g = inv * gamma[64 + m] * sc;
    for (int j = 0; j < 5; ++j) { int idx = 128 + m * 5 + j; p[idx] = p[idx] * g; }
  }
}

extern "C" void kernel_launch(void* const* d_in, const int* in_sizes, int n_in,
                              void* d_out, int out_size, void* d_ws, size_t ws_size,
                              hipStream_t stream) {
  (void)n_in; (void)out_size; (void)ws_size;
  const int*   edge_index   = (const int*)d_in[0];
  const float* edge_sh      = (const float*)d_in[3];
  const float* edge_length  = (const float*)d_in[4];
  const float* edge_one_hot = (const float*)d_in[5];
  const float* bessel_w     = (const float*)d_in[6];
  const float* W1    = (const float*)d_in[7];
  const float* W2    = (const float*)d_in[8];
  const float* W3    = (const float*)d_in[9];
  const float* W_env = (const float*)d_in[10];
  const float* gamma = (const float*)d_in[11];
  const float* beta  = (const float*)d_in[12];

  const int E = in_sizes[4];   // edge_length element count
  const int N = in_sizes[1];   // atom_type element count

  float* out = (float*)d_out;
  const long long latOff  = 0;
  const long long nodeOff = (long long)E * 128;
  const long long efOff   = nodeOff + (long long)N * 288;
  const long long cutOff  = efOff + (long long)E * 288;
  const long long maskOff = cutOff + (long long)E;

  _Float16* wts = (_Float16*)d_ws;

  // zero the node accumulator region of d_out (edge kernel atomically adds into it)
  hipMemsetAsync(out + nodeOff, 0, (size_t)N * 288 * sizeof(float), stream);

  convert_weights<<<(WTS_TOTAL + 255) / 256, 256, 0, stream>>>(W1, W2, W3, W_env, wts);

  const int numTiles = (E + 15) / 16;
  edge_kernel<<<2048, 256, 0, stream>>>(edge_index, edge_sh, edge_length, edge_one_hot,
                                        bessel_w, wts,
                                        out + latOff, out + nodeOff, out + efOff,
                                        out + cutOff, out + maskOff, E, numTiles);

  node_kernel<<<(N + 255) / 256, 256, 0, stream>>>(out + nodeOff, gamma, beta, N);
}
